// BERT_22187801051543
// MI455X (gfx1250) — compile-verified, matrix operand order checked by reading, not run on hardware
//
#include <hip/hip_runtime.h>
#include <hip/hip_bf16.h>
#include <math.h>

// ---------------- types for WMMA ----------------
typedef __attribute__((ext_vector_type(16))) __bf16 v16bf;
typedef __attribute__((ext_vector_type(8)))  __bf16 v8bf;
typedef __attribute__((ext_vector_type(8)))  float  v8f;

// ---------------- model dims ----------------
#define D_MODEL 144
#define NHEADS  6
#define DK      24
#define SEQ     60
#define BATCH   128
#define DFF     576
#define NMASK   20
#define NTRI    64
#define VOCAB_N 32000
#define NLAYERS 6
#define ROWS    (BATCH*SEQ)    // 7680
#define MROWS   (BATCH*NMASK)  // 2560

// ---------------- helpers ----------------
__device__ inline float wave_sum(float v) {
#pragma unroll
  for (int off = 16; off > 0; off >>= 1) v += __shfl_xor(v, off, 32);
  return v;
}
__device__ inline float gelu_f(float x) {
  return x * 0.5f * (1.0f + erff(x * 0.70710678118654752f));
}

// bf16 WMMA fragment: elements 0..7 = 8 contiguous values at p, 8..15 = at p+16.
// (A: M over lanes 0-15, khalf folded into p by caller. B: N over lanes, same packing.)
__device__ inline v16bf ldfrag(const __bf16* __restrict__ p) {
  v8bf lo = *reinterpret_cast<const v8bf*>(p);
  v8bf hi = *reinterpret_cast<const v8bf*>(p + 16);
  return __builtin_shufflevector(lo, hi, 0,1,2,3,4,5,6,7,8,9,10,11,12,13,14,15);
}
__device__ inline v16bf ldfrag_tail(const __bf16* __restrict__ p) {  // K-tail: upper half = 0
  v8bf lo = *reinterpret_cast<const v8bf*>(p);
  v16bf r;
#pragma unroll
  for (int i = 0; i < 8; i++) { r[i] = lo[i]; r[i+8] = (__bf16)0.f; }
  return r;
}

// ---------------- f32 -> bf16 pre-conversion ----------------
__global__ void k_cvt_bf16(const float* __restrict__ src, __bf16* __restrict__ dst, long n) {
  long i = blockIdx.x * (long)blockDim.x + threadIdx.x;
  long stride = (long)gridDim.x * blockDim.x;
  for (; i < n; i += stride) dst[i] = (__bf16)src[i];
}

// ---------------- generic GEMM: C[M,N] = A[M,K] * W[N,K]^T + bias, opt GELU ----------------
// A and W both pre-converted bf16 (pure fragment loads, zero cvt in the K-loop).
// One wave computes a 64x16 output strip (4 M-subtiles reuse each B fragment).
// Epilogue: optional f32 store (C) and/or bf16 mirror (Cb); rowdiv>1 keeps only rows
// with row%rowdiv==0 in the mirror, stored compacted at row/rowdiv.
// Requirements: M % 64 == 0, N % 16 == 0, K % 16 == 0. All call sites satisfy this.
__global__ void k_gemm_bf16(const __bf16* __restrict__ A, long lda,
                            const __bf16* __restrict__ W, long ldw,
                            const float* __restrict__ bias,
                            float* __restrict__ C, long ldc,
                            __bf16* __restrict__ Cb, long ldcb, int rowdiv,
                            int M, int N, int K, int act) {
  int wid  = (int)((blockIdx.x * (long)blockDim.x + threadIdx.x) >> 5);
  int lane = threadIdx.x & 31;
  int tilesN = N >> 4;
  int total  = (M >> 6) * tilesN;
  if (wid >= total) return;                 // wave-uniform exit (EXEC all-1 for WMMA)
  int tm = wid / tilesN, tn = wid % tilesN;

  int rowA  = (tm << 6) + (lane & 15);
  int rowW  = (tn << 4) + (lane & 15);
  int khalf = (lane & 16) ? 8 : 0;          // upper 16 lanes cover K+8 / K+24 sub-columns
  const __bf16* Ap0 = A + (size_t)rowA * lda + khalf;
  const __bf16* Ap1 = Ap0 + (size_t)16 * lda;
  const __bf16* Ap2 = Ap1 + (size_t)16 * lda;
  const __bf16* Ap3 = Ap2 + (size_t)16 * lda;
  const __bf16* Wp  = W + (size_t)rowW * ldw + khalf;

  v8f acc0 = (v8f){0.f,0.f,0.f,0.f,0.f,0.f,0.f,0.f};
  v8f acc1 = acc0, acc2 = acc0, acc3 = acc0;

  int kfull = K & ~31;
  for (int k0 = 0; k0 < kfull; k0 += 32) {
    v16bf b  = ldfrag(Wp  + k0);
    v16bf a0 = ldfrag(Ap0 + k0);
    v16bf a1 = ldfrag(Ap1 + k0);
    v16bf a2 = ldfrag(Ap2 + k0);
    v16bf a3 = ldfrag(Ap3 + k0);
    acc0 = __builtin_amdgcn_wmma_f32_16x16x32_bf16(false, a0, false, b, (short)0, acc0, false, false);
    acc1 = __builtin_amdgcn_wmma_f32_16x16x32_bf16(false, a1, false, b, (short)0, acc1, false, false);
    acc2 = __builtin_amdgcn_wmma_f32_16x16x32_bf16(false, a2, false, b, (short)0, acc2, false, false);
    acc3 = __builtin_amdgcn_wmma_f32_16x16x32_bf16(false, a3, false, b, (short)0, acc3, false, false);
  }
  if (K & 16) {                             // 16-wide K tail (e.g. K=144)
    v16bf b  = ldfrag_tail(Wp  + kfull);
    v16bf a0 = ldfrag_tail(Ap0 + kfull);
    v16bf a1 = ldfrag_tail(Ap1 + kfull);
    v16bf a2 = ldfrag_tail(Ap2 + kfull);
    v16bf a3 = ldfrag_tail(Ap3 + kfull);
    acc0 = __builtin_amdgcn_wmma_f32_16x16x32_bf16(false, a0, false, b, (short)0, acc0, false, false);
    acc1 = __builtin_amdgcn_wmma_f32_16x16x32_bf16(false, a1, false, b, (short)0, acc1, false, false);
    acc2 = __builtin_amdgcn_wmma_f32_16x16x32_bf16(false, a2, false, b, (short)0, acc2, false, false);
    acc3 = __builtin_amdgcn_wmma_f32_16x16x32_bf16(false, a3, false, b, (short)0, acc3, false, false);
  }

  int col = (tn << 4) + (lane & 15);
  int r0  = (tm << 6) + ((lane & 16) ? 8 : 0);
  float bv = bias ? bias[col] : 0.f;
  const v8f* accs[4] = {&acc0, &acc1, &acc2, &acc3};
#pragma unroll
  for (int mi = 0; mi < 4; mi++) {
#pragma unroll
    for (int r = 0; r < 8; r++) {
      int row = r0 + mi*16 + r;
      float v = (*accs[mi])[r] + bv;
      if (act == 1) v = gelu_f(v);
      if (C) C[(size_t)row * ldc + col] = v;
      if (Cb) {
        if (rowdiv == 1)            Cb[(size_t)row * ldcb + col] = (__bf16)v;
        else if (row % rowdiv == 0) Cb[(size_t)(row / rowdiv) * ldcb + col] = (__bf16)v;
      }
    }
  }
}

// ---------------- embedding + layernorm (one wave per row; f32 + bf16 outputs) ----------------
__global__ void k_embed_ln(const int* __restrict__ ids, const float* __restrict__ te,
                           const float* __restrict__ pe, const float* __restrict__ g,
                           const float* __restrict__ bb, float* __restrict__ y,
                           __bf16* __restrict__ yb) {
  int wid  = (int)((blockIdx.x * (long)blockDim.x + threadIdx.x) >> 5);
  int lane = threadIdx.x & 31;
  if (wid >= ROWS) return;
  int tok = ids[wid];
  int sp  = wid % SEQ;
  const float* tp = te + (size_t)tok * D_MODEL;
  const float* pp = pe + (size_t)sp  * D_MODEL;
  float v[5]; float s = 0.f, s2 = 0.f;
#pragma unroll
  for (int j = 0; j < 5; j++) {
    int idx = lane + 32*j;
    float t = 0.f;
    if (idx < D_MODEL) t = tp[idx] + pp[idx];
    v[j] = t; s += t; s2 += t*t;
  }
  s = wave_sum(s); s2 = wave_sum(s2);
  float mu  = s  * (1.f/D_MODEL);
  float var = s2 * (1.f/D_MODEL) - mu*mu;
  float inv = rsqrtf(var + 1e-5f);
  float*  yp  = y  + (size_t)wid * D_MODEL;
  __bf16* ybp = yb + (size_t)wid * D_MODEL;
#pragma unroll
  for (int j = 0; j < 5; j++) {
    int idx = lane + 32*j;
    if (idx < D_MODEL) {
      float o = (v[j]-mu)*inv*g[idx] + bb[idx];
      yp[idx] = o; ybp[idx] = (__bf16)o;
    }
  }
}

// ---------------- layernorm of (x + res) -> bf16 (feeds fc1 GEMM only) ----------------
__global__ void k_ln(const float* __restrict__ x, const float* __restrict__ res,
                     const float* __restrict__ g, const float* __restrict__ bb,
                     __bf16* __restrict__ yb, int rows) {
  int wid  = (int)((blockIdx.x * (long)blockDim.x + threadIdx.x) >> 5);
  int lane = threadIdx.x & 31;
  if (wid >= rows) return;
  const float* xp = x   + (size_t)wid * D_MODEL;
  const float* rp = res + (size_t)wid * D_MODEL;
  float v[5]; float s = 0.f, s2 = 0.f;
#pragma unroll
  for (int j = 0; j < 5; j++) {
    int idx = lane + 32*j;
    float t = 0.f;
    if (idx < D_MODEL) t = xp[idx] + rp[idx];
    v[j] = t; s += t; s2 += t*t;
  }
  s = wave_sum(s); s2 = wave_sum(s2);
  float mu  = s  * (1.f/D_MODEL);
  float var = s2 * (1.f/D_MODEL) - mu*mu;
  float inv = rsqrtf(var + 1e-5f);
  __bf16* yp = yb + (size_t)wid * D_MODEL;
#pragma unroll
  for (int j = 0; j < 5; j++) {
    int idx = lane + 32*j;
    if (idx < D_MODEL) yp[idx] = (__bf16)((v[j]-mu)*inv*g[idx] + bb[idx]);
  }
}

// ---------------- attention: tanh(0.1 * QK^T/sqrt(24)) @ V -> bf16 ctx ----------------
__global__ void k_attention(const float* __restrict__ Q, const float* __restrict__ K,
                            const float* __restrict__ V, __bf16* __restrict__ O) {
  __shared__ float Ks[SEQ*DK];
  __shared__ float Vs[SEQ*DK];
  int bh = blockIdx.x;
  int b = bh / NHEADS, h = bh % NHEADS;
  const float* Kb = K + ((size_t)b*SEQ)*D_MODEL + h*DK;
  const float* Vb = V + ((size_t)b*SEQ)*D_MODEL + h*DK;
  for (int i = threadIdx.x; i < SEQ*DK; i += blockDim.x) {
    int s = i / DK, d = i % DK;
    Ks[i] = Kb[(size_t)s*D_MODEL + d];
    Vs[i] = Vb[(size_t)s*D_MODEL + d];
  }
  __syncthreads();
  int s = threadIdx.x;
  if (s < SEQ) {
    const float* qp = Q + ((size_t)b*SEQ + s)*D_MODEL + h*DK;
    float q[DK];
#pragma unroll
    for (int d = 0; d < DK; d++) q[d] = qp[d];
    float acc[DK] = {};
    const float scale = 0.1f * 0.20412414523193154f;   // 0.1 / sqrt(24)
    for (int j = 0; j < SEQ; j++) {
      float sc = 0.f;
#pragma unroll
      for (int d = 0; d < DK; d++) sc += q[d] * Ks[j*DK + d];
      float a = tanhf(sc * scale);
#pragma unroll
      for (int d = 0; d < DK; d++) acc[d] += a * Vs[j*DK + d];
    }
    __bf16* op = O + ((size_t)b*SEQ + s)*D_MODEL + h*DK;
#pragma unroll
    for (int d = 0; d < DK; d++) op[d] = (__bf16)acc[d];
  }
}

// ---------------- gather masked rows -> bf16 (feeds lin GEMM only) ----------------
__global__ void k_gather(const float* __restrict__ X, const int* __restrict__ mp,
                         __bf16* __restrict__ Hm) {
  int i = blockIdx.x * blockDim.x + threadIdx.x;
  if (i >= MROWS * D_MODEL) return;
  int row = i / D_MODEL, col = i % D_MODEL;
  int b = row / NMASK, m = row % NMASK;
  int s = mp[b*NMASK + m];
  Hm[i] = (__bf16)X[((size_t)b*SEQ + s)*D_MODEL + col];
}

// ---------------- tri-embedding compression (one wave per (b,t,c)) ----------------
__global__ void k_tri_compress(const int* __restrict__ tri, const float* __restrict__ te,
                               const float* __restrict__ w, const float* __restrict__ bb,
                               float* __restrict__ o_tri, float* __restrict__ o_corr) {
  int wid  = (int)((blockIdx.x * (long)blockDim.x + threadIdx.x) >> 5);
  int lane = threadIdx.x & 31;
  if (wid >= BATCH*NTRI*3) return;
  int tok = tri[wid];
  const float* vp = te + (size_t)tok * D_MODEL;
  float s0 = 0.f, s1 = 0.f;
  for (int idx = lane; idx < D_MODEL; idx += 32) {
    float v = vp[idx];
    s0 += v * w[idx];
    s1 += v * w[D_MODEL + idx];
  }
  s0 = wave_sum(s0); s1 = wave_sum(s1);
  if (lane == 0) {
    float a = s0 + bb[0], c = s1 + bb[1];
    o_tri [wid*2+0] = a; o_tri [wid*2+1] = c;
    o_corr[wid*2+0] = a; o_corr[wid*2+1] = c;
  }
}

struct GraphP {
  const float* w0; const float* b0;
  const float* w1; const float* b1;
  const float* w2; const float* b2;
};

// ---------------- graph attention row-0 weights: a0[g,t] = tanh(0.1 * Query0 . ce(g,0,t)) ----------------
__global__ void k_graph_a0(const int* __restrict__ tri, const float* __restrict__ te,
                           const float* __restrict__ q0, GraphP gp, float* __restrict__ a0) {
  int wid  = (int)((blockIdx.x * (long)blockDim.x + threadIdx.x) >> 5);
  int lane = threadIdx.x & 31;
  if (wid >= 3*NTRI) return;
  int g = wid / NTRI, t = wid % NTRI;
  const float* gw = (g == 0) ? gp.w0 : ((g == 1) ? gp.w1 : gp.w2);
  const float* gb = (g == 0) ? gp.b0 : ((g == 1) ? gp.b1 : gp.b2);
  int e0 = tri[t*3+0], e1 = tri[t*3+1], e2 = tri[t*3+2];   // batch 0
  float w0 = gw[0], w1 = gw[1], w2 = gw[2], bi = gb[0];
  float s = 0.f;
  for (int d = lane; d < D_MODEL; d += 32) {
    float ce = te[(size_t)e0*D_MODEL+d]*w0 + te[(size_t)e1*D_MODEL+d]*w1
             + te[(size_t)e2*D_MODEL+d]*w2 + bi;
    s += q0[d] * ce;
  }
  s = wave_sum(s);
  if (lane == 0) a0[wid] = tanhf(0.1f * s);
}

// ---------------- graph context + compress: temp2[b,d] ----------------
__global__ void k_graph_ctx(const int* __restrict__ tri, const float* __restrict__ te,
                            const float* __restrict__ a0, GraphP gp,
                            const float* __restrict__ cw, const float* __restrict__ cb,
                            float* __restrict__ t2) {
  int b = blockIdx.x;
  int d = threadIdx.x;
  if (d >= D_MODEL) return;
  float w[3][3], bi[3];
  const float* gws[3] = {gp.w0, gp.w1, gp.w2};
  const float* gbs[3] = {gp.b0, gp.b1, gp.b2};
#pragma unroll
  for (int g = 0; g < 3; g++) {
#pragma unroll
    for (int c = 0; c < 3; c++) w[g][c] = gws[g][c];
    bi[g] = gbs[g][0];
  }
  float acc[3] = {0.f, 0.f, 0.f};
  for (int t = 0; t < NTRI; t++) {
    int e0 = tri[(b*NTRI+t)*3+0], e1 = tri[(b*NTRI+t)*3+1], e2 = tri[(b*NTRI+t)*3+2];
    float v0 = te[(size_t)e0*D_MODEL+d];
    float v1 = te[(size_t)e1*D_MODEL+d];
    float v2 = te[(size_t)e2*D_MODEL+d];
#pragma unroll
    for (int g = 0; g < 3; g++) {
      float ce = v0*w[g][0] + v1*w[g][1] + v2*w[g][2] + bi[g];
      acc[g] += a0[g*NTRI + t] * ce;
    }
  }
  t2[b*D_MODEL + d] = acc[0]*cw[0] + acc[1]*cw[1] + acc[2]*cw[2] + cb[0];
}

// ---------------- final prediction head (one wave per batch row) ----------------
__global__ void k_pred(const float* __restrict__ query, const float* __restrict__ t2,
                       const float* __restrict__ w, const float* __restrict__ bb,
                       float* __restrict__ out) {
  int b    = (int)((blockIdx.x * (long)blockDim.x + threadIdx.x) >> 5);
  int lane = threadIdx.x & 31;
  if (b >= BATCH) return;
  float s0 = 0.f, s1 = 0.f;
  for (int i = lane; i < 2*D_MODEL; i += 32) {
    float x = (i < D_MODEL) ? query[b*D_MODEL + i] : t2[b*D_MODEL + (i - D_MODEL)];
    s0 += x * w[i];
    s1 += x * w[2*D_MODEL + i];
  }
  s0 = wave_sum(s0); s1 = wave_sum(s1);
  if (lane == 0) { out[b*2+0] = s0 + bb[0]; out[b*2+1] = s1 + bb[1]; }
}

// ---------------- host side ----------------
static inline void gemm(const __bf16* A, long lda, const __bf16* W, long ldw,
                        const float* bias, float* C, long ldc,
                        __bf16* Cb, long ldcb, int rowdiv,
                        int M, int N, int K, int act, hipStream_t stream) {
  int total  = (M/64) * (N/16);
  int blocks = (total + 7) / 8;          // 8 waves (256 threads) per block
  k_gemm_bf16<<<blocks, 256, 0, stream>>>(A, lda, W, ldw, bias, C, ldc,
                                          Cb, ldcb, rowdiv, M, N, K, act);
}

static inline void cvt(const float* s, __bf16* d, long n, hipStream_t stream) {
  long blocks = (n + 255) / 256;
  if (blocks > 4096) blocks = 4096;
  k_cvt_bf16<<<(int)blocks, 256, 0, stream>>>(s, d, n);
}

extern "C" void kernel_launch(void* const* d_in, const int* in_sizes, int n_in,
                              void* d_out, int out_size, void* d_ws, size_t ws_size,
                              hipStream_t stream) {
  (void)in_sizes; (void)n_in; (void)out_size; (void)ws_size;
#define PF(i) ((const float*)d_in[i])
  const int* input_ids  = (const int*)d_in[0];
  const int* masked_pos = (const int*)d_in[1];
  const int* emb_tri    = (const int*)d_in[2];
  // d_in[3] = corr_tri (unused by the reference's effective math)
  const float* tok_embed = PF(4);
  const float* pos_embed = PF(5);
  const float* eln_g = PF(6), *eln_b = PF(7);
  const float* wk_b = PF(9);
  const float* wv_b = PF(11);
  // layers: base index 12 + 10*l : {W_Q.w, W_Q.b, attn_out.w, attn_out.b, ln.g, ln.b, fc1.w, fc1.b, fc2.w, fc2.b}
  const float* lin_b = PF(73);
  const float* d1_b  = PF(75);
  const float* d2_w  = PF(76), *d2_b = PF(77);
  GraphP gp{ PF(78), PF(79), PF(80), PF(81), PF(82), PF(83) };
  const float* cw = PF(84), *cb = PF(85);
  const float* tc_w = PF(86), *tc_b = PF(87);

  // ---------------- workspace layout ----------------
  // f32 activations
  float* ws = (float*)d_ws;
  const size_t SZX = (size_t)ROWS * D_MODEL;      // 1,105,920
  float* X   = ws;  ws += SZX;                    // current residual stream (f32)
  float* Qb  = ws;  ws += SZX;
  float* Kb  = ws;  ws += SZX;
  float* Vb  = ws;  ws += SZX;
  float* T1  = ws;  ws += SZX;
  float* a0  = ws;  ws += 192;
  float* t2  = ws;  ws += (size_t)BATCH * D_MODEL;   // total f32: 5,548,224 (16B-aligned end)

  // bf16 activations (GEMM A operands)
  __bf16* wb = (__bf16*)ws;
  __bf16* Xb16   = wb;  wb += SZX;                        // mirror of X
  __bf16* Cxb16  = wb;  wb += SZX;                        // attention context
  __bf16* X2b16  = wb;  wb += SZX;                        // post-LN
  __bf16* FFb16  = wb;  wb += (size_t)ROWS * DFF;         // fc1 output (GELU'd)
  __bf16* Hmb16  = wb;  wb += (size_t)MROWS * D_MODEL;
  __bf16* Hgb16  = wb;  wb += (size_t)MROWS * D_MODEL;
  __bf16* Qrow16 = wb;  wb += (size_t)BATCH * VOCAB_N;    // logits rows m==0, compacted

  // bf16 weight copies (all sizes multiples of 8 -> 16B alignment preserved)
  const size_t SZW = (size_t)D_MODEL * D_MODEL;   // 20736
  __bf16* TEb = wb;  wb += (size_t)VOCAB_N * D_MODEL;     // 4,608,000
  __bf16* WKb = wb;  wb += SZW;
  __bf16* WVb = wb;  wb += SZW;
  __bf16* WQb[NLAYERS]; __bf16* AOb[NLAYERS]; __bf16* F1b[NLAYERS]; __bf16* F2b[NLAYERS];
  for (int l = 0; l < NLAYERS; l++) {
    WQb[l] = wb; wb += SZW;
    AOb[l] = wb; wb += SZW;
    F1b[l] = wb; wb += (size_t)DFF * D_MODEL;     // 82,944
    F2b[l] = wb; wb += (size_t)D_MODEL * DFF;     // 82,944
  }
  __bf16* LINb = wb; wb += SZW;
  __bf16* D1b  = wb; wb += (size_t)D_MODEL * VOCAB_N;     // 4,608,000

  // ---------------- output layout ----------------
  float* out     = (float*)d_out;
  float* o_lm    = out;                                   // 128*20*32000
  float* o_pred  = out + (size_t)MROWS * VOCAB_N;         // 128*2
  float* o_query = o_pred + (size_t)BATCH * 2;            // 128*144
  float* o_tri   = o_query + (size_t)BATCH * D_MODEL;     // 128*64*3*2
  float* o_corr  = o_tri + (size_t)BATCH * NTRI * 3 * 2;  // 128*64*3*2

  // 0) pre-convert all GEMM weights to bf16
  cvt(tok_embed, TEb, (long)VOCAB_N * D_MODEL, stream);
  cvt(PF(8),  WKb, SZW, stream);
  cvt(PF(10), WVb, SZW, stream);
  for (int l = 0; l < NLAYERS; l++) {
    int base = 12 + 10*l;
    cvt(PF(base+0), WQb[l], SZW, stream);
    cvt(PF(base+2), AOb[l], SZW, stream);
    cvt(PF(base+6), F1b[l], (long)DFF * D_MODEL, stream);
    cvt(PF(base+8), F2b[l], (long)D_MODEL * DFF, stream);
  }
  cvt(PF(72), LINb, SZW, stream);
  cvt(PF(74), D1b, (long)D_MODEL * VOCAB_N, stream);

  // 1) embedding + LN -> X (f32) + Xb16
  k_embed_ln<<<ROWS/8, 256, 0, stream>>>(input_ids, tok_embed, pos_embed, eln_g, eln_b, X, Xb16);

  // 2) K and V, computed once from embedding output (as in source)
  gemm(Xb16, D_MODEL, WKb, D_MODEL, wk_b, Kb, D_MODEL, nullptr, 0, 1,
       ROWS, D_MODEL, D_MODEL, 0, stream);
  gemm(Xb16, D_MODEL, WVb, D_MODEL, wv_b, Vb, D_MODEL, nullptr, 0, 1,
       ROWS, D_MODEL, D_MODEL, 0, stream);

  // 3) transformer layers
  for (int l = 0; l < NLAYERS; l++) {
    int base = 12 + 10*l;
    const float* wq_b = PF(base+1);
    const float* ao_b = PF(base+3);
    const float* lng  = PF(base+4); const float* lnb = PF(base+5);
    const float* f1b  = PF(base+7);
    const float* f2b  = PF(base+9);
    gemm(Xb16, D_MODEL, WQb[l], D_MODEL, wq_b, Qb, D_MODEL, nullptr, 0, 1,
         ROWS, D_MODEL, D_MODEL, 0, stream);
    k_attention<<<BATCH*NHEADS, 64, 0, stream>>>(Qb, Kb, Vb, Cxb16);
    gemm(Cxb16, D_MODEL, AOb[l], D_MODEL, ao_b, T1, D_MODEL, nullptr, 0, 1,
         ROWS, D_MODEL, D_MODEL, 0, stream);
    k_ln<<<ROWS/8, 256, 0, stream>>>(T1, X, lng, lnb, X2b16, ROWS);
    gemm(X2b16, D_MODEL, F1b[l], D_MODEL, f1b, nullptr, 0, FFb16, DFF, 1,
         ROWS, DFF, D_MODEL, 1, stream);                                   // GELU, bf16-only out
    gemm(FFb16, DFF, F2b[l], DFF, f2b, X, D_MODEL, Xb16, D_MODEL, 1,
         ROWS, D_MODEL, DFF, 0, stream);                                   // f32 X + bf16 mirror
  }

  // 4) masked gather -> GELU linear -> logits_lm = Hg @ tok_embed^T (tied, no bias)
  k_gather<<<(MROWS*D_MODEL + 255)/256, 256, 0, stream>>>(X, masked_pos, Hmb16);
  gemm(Hmb16, D_MODEL, LINb, D_MODEL, lin_b, nullptr, 0, Hgb16, D_MODEL, 1,
       MROWS, D_MODEL, D_MODEL, 1, stream);                                // GELU, bf16-only out
  gemm(Hgb16, D_MODEL, TEb, D_MODEL, nullptr, o_lm, VOCAB_N, Qrow16, VOCAB_N, NMASK,
       MROWS, VOCAB_N, D_MODEL, 0, stream);   // f32 logits + compact bf16 mirror of m==0 rows

  // 5) Query = logits_lm[:,0,:] @ dense1.w^T + b
  gemm(Qrow16, VOCAB_N, D1b, VOCAB_N, d1_b, o_query, D_MODEL, nullptr, 0, 1,
       BATCH, D_MODEL, VOCAB_N, 0, stream);

  // 6) tri compression (written to both tri and corr outputs)
  k_tri_compress<<<(BATCH*NTRI*3)/8, 256, 0, stream>>>(emb_tri, tok_embed, tc_w, tc_b, o_tri, o_corr);

  // 7) graph heads (batch-0 broadcast semantics) + prediction head
  k_graph_a0<<<(3*NTRI)/8, 256, 0, stream>>>(emb_tri, tok_embed, o_query, gp, a0);
  k_graph_ctx<<<BATCH, 160, 0, stream>>>(emb_tri, tok_embed, a0, gp, cw, cb, t2);
  k_pred<<<BATCH/8, 256, 0, stream>>>(o_query, t2, d2_w, d2_b, o_pred);
#undef PF
}
